// TrivialTreeTagger_31267361914882
// MI455X (gfx1250) — compile-verified, matrix operand order checked by reading, not run on hardware
//
#include <hip/hip_runtime.h>
#include <hip/hip_bf16.h>
#include <math.h>

// ---------------------------------------------------------------------------
// TrivialTreeTagger on MI455X (gfx1250), wave32 + v_wmma_f32_16x16x32_f16
// + global_load_async_to_lds_b128 for activation staging.
//
//  * First LSTM step is node-independent -> (h1,c1) precomputed once.
//  * Leaf output depends only on ident (26 values) -> 26-row table + gather.
//  * f16-input / f32-accumulate WMMA; weights pre-packed into B-fragment
//    layout; h / combined / scores kept f16 (they only feed f16 WMMA).
//  * 256-thread blocks (8 wave32s), 4 gate tiles per wave -> <256 VGPRs.
// ---------------------------------------------------------------------------

typedef __attribute__((ext_vector_type(16))) _Float16 v16h;
typedef __attribute__((ext_vector_type(8)))  float    v8f;

#define WMMA_F16(a, b, c) \
  __builtin_amdgcn_wmma_f32_16x16x32_f16(false, (a), false, (b), (short)0, (c), false, false)

// K index held by (lane, half t) in a 16-bit A/B fragment (16x32 / 32x16).
__device__ __forceinline__ int kmap(int lane, int t) {
  return ((t >> 3) << 4) + ((lane >> 4) << 3) + (t & 7);
}

__device__ __forceinline__ v16h frag_from_f16_lds(const _Float16* base, int ld,
                                                  int kbase, int lane) {
  const int m = lane & 15;
  v16h a;
#pragma unroll
  for (int t = 0; t < 16; ++t) a[t] = base[m * ld + kbase + kmap(lane, t)];
  return a;
}

__device__ __forceinline__ v16h load_bfrag(const _Float16* buf, int tile, int lane) {
  return *reinterpret_cast<const v16h*>(buf + (((size_t)tile * 32 + lane) << 4));
}

__device__ __forceinline__ float sigmoidf_(float x) {
  return 1.0f / (1.0f + __expf(-x));
}

// ---------------------------------------------------------------------------
// Prep 1: bias_sum = b_ih + b_hh ; (h1, c1) = first LSTM step (constant).
// ---------------------------------------------------------------------------
__global__ __launch_bounds__(512) void prep_const_kernel(
    const float* __restrict__ W_ih, const float* __restrict__ b_ih,
    const float* __restrict__ b_hh, const float* __restrict__ lstm_init,
    float* __restrict__ bias_sum, float* __restrict__ h1,
    float* __restrict__ c1) {
  __shared__ float g0[512];
  __shared__ float xin[128];
  const int t = threadIdx.x;
  if (t < 128) xin[t] = lstm_init[t];
  __syncthreads();
  float b = b_ih[t] + b_hh[t];
  bias_sum[t] = b;
  const float* row = W_ih + t * 128;
  float acc = b;
#pragma unroll 4
  for (int k = 0; k < 128; ++k) acc += row[k] * xin[k];
  g0[t] = acc;
  __syncthreads();
  if (t < 128) {
    float si = sigmoidf_(g0[t]);        // input gate
    float cg = tanhf(g0[256 + t]);      // candidate
    float cc = si * cg;                 // c = 0 so forget path vanishes
    float so = sigmoidf_(g0[384 + t]);  // output gate
    c1[t] = cc;
    h1[t] = so * tanhf(cc);
  }
}

// ---------------------------------------------------------------------------
// Prep 2: pack W_ih^T, W_hh^T, W_comb^T, W_tag^T into f16 WMMA B fragments.
// ---------------------------------------------------------------------------
__global__ __launch_bounds__(32) void prep_weights_kernel(
    const float* __restrict__ W_ih, const float* __restrict__ W_hh,
    const float* __restrict__ W_comb, const float* __restrict__ W_tag,
    _Float16* __restrict__ Bih, _Float16* __restrict__ Bhh,
    _Float16* __restrict__ Bcomb, _Float16* __restrict__ Btag) {
  const int tile = blockIdx.x;
  const int lane = threadIdx.x;
  const int n = lane & 15;

  const float* W;
  _Float16* dstbuf;
  int localTile, j0, k0, ldk;
  if (tile < 128) {
    W = W_ih; dstbuf = Bih; localTile = tile;
    j0 = localTile >> 2; k0 = localTile & 3; ldk = 128;
  } else if (tile < 256) {
    W = W_hh; dstbuf = Bhh; localTile = tile - 128;
    j0 = localTile >> 2; k0 = localTile & 3; ldk = 128;
  } else if (tile < 320) {
    W = W_comb; dstbuf = Bcomb; localTile = tile - 256;
    j0 = localTile >> 3; k0 = localTile & 7; ldk = 256;
  } else {
    W = W_tag; dstbuf = Btag; localTile = tile - 320;
    j0 = localTile >> 2; k0 = localTile & 3; ldk = 128;
  }
  const float* row = W + (size_t)(j0 * 16 + n) * ldk + k0 * 32;
  _Float16* dp = dstbuf + (((size_t)localTile * 32 + lane) << 4);
#pragma unroll
  for (int t = 0; t < 16; ++t) dp[t] = (_Float16)row[kmap(lane, t)];
}

// ---------------------------------------------------------------------------
// Level kernel: 16 nodes per block, 256 threads (8 wave32s).
// ---------------------------------------------------------------------------
__global__ __launch_bounds__(256) void level_kernel(
    const _Float16* __restrict__ scores_prev, _Float16* __restrict__ scores_out,
    float* __restrict__ out_f32, int N, const int* __restrict__ idents,
    const _Float16* __restrict__ Bih, const _Float16* __restrict__ Bhh,
    const _Float16* __restrict__ Bcomb, const _Float16* __restrict__ Btag,
    const float* __restrict__ bias_sum, const float* __restrict__ h1,
    const float* __restrict__ c1, const float* __restrict__ emb_table,
    const float* __restrict__ b_comb, const float* __restrict__ b_tag,
    int has_children, int use_iota) {
  __shared__ __align__(16) float    sm_g[16 * 512];  // gates; later: tag [0,2048)
  __shared__ __align__(16) _Float16 sm_h[16 * 128];  // f16: feeds WMMA only
  __shared__ __align__(16) float    sm_c[16 * 128];
  __shared__ __align__(16) _Float16 sm_x[16 * 128];  // x stage; later: combined
  __shared__ int sm_ident[16];

  const int tid  = threadIdx.x;
  const int lane = tid & 31;
  const int wave = tid >> 5;  // 0..7
  const int base = blockIdx.x << 4;

  for (int e = tid; e < 2048; e += 256) {
    sm_h[e] = (_Float16)h1[e & 127];
    sm_c[e] = c1[e & 127];
  }
  if (tid < 16) {
    int node = base + tid;
    int nc = (node < N) ? node : 0;
    sm_ident[tid] = use_iota ? nc : idents[nc];
  }
  __syncthreads();

  if (has_children) {
    const unsigned lds_x = (unsigned)(uintptr_t)sm_x;  // LDS byte offset
#pragma unroll 1
    for (int s = 0; s < 2; ++s) {
      // Async-stage x = scores_prev[2*node + s] into LDS: one 16 B chunk per
      // thread (256 * 16 B == full 16x128 f16 tile), ASYNCcnt-tracked.
      {
        const int m = tid >> 4;        // row 0..15
        const int c16 = tid & 15;      // 16 B chunk within row
        const size_t row = (size_t)2 * (base + m) + s;
        const _Float16* gp = scores_prev + row * 128 + c16 * 8;
        const unsigned ldst = lds_x + (unsigned)tid * 16u;
        asm volatile("global_load_async_to_lds_b128 %0, %1, off"
                     :: "v"(ldst), "v"(gp)
                     : "memory");
      }
      asm volatile("s_wait_asynccnt 0x0" ::: "memory");
      __syncthreads();

      // g = x @ W_ih^T + h @ W_hh^T (16 x 512); wave w owns tiles j0=4w..4w+3
      v16h ax[4], ah[4];
#pragma unroll
      for (int k0 = 0; k0 < 4; ++k0) {
        ax[k0] = frag_from_f16_lds(sm_x, 128, k0 * 32, lane);
        ah[k0] = frag_from_f16_lds(sm_h, 128, k0 * 32, lane);
      }
      v8f acc[4] = {};
#pragma unroll
      for (int t = 0; t < 4; ++t) {
        int j0 = wave * 4 + t;
#pragma unroll
        for (int k0 = 0; k0 < 4; ++k0) {
          acc[t] = WMMA_F16(ax[k0], load_bfrag(Bih, j0 * 4 + k0, lane), acc[t]);
          acc[t] = WMMA_F16(ah[k0], load_bfrag(Bhh, j0 * 4 + k0, lane), acc[t]);
        }
      }
      // Spill gate tiles (C layout: col = lane&15, rows by vgpr index).
      {
        int col = lane & 15;
        int mb = (lane >> 4) << 3;
#pragma unroll
        for (int t = 0; t < 4; ++t) {
          int j0 = wave * 4 + t;
#pragma unroll
          for (int r = 0; r < 8; ++r)
            sm_g[(mb + r) * 512 + j0 * 16 + col] = acc[t][r];
        }
      }
      __syncthreads();

      // Elementwise LSTM update (i, f, g, o order per jnp.split).
      for (int e = tid; e < 2048; e += 256) {
        int m = e >> 7, j = e & 127;
        float gi = sm_g[m * 512 + j]       + bias_sum[j];
        float gf = sm_g[m * 512 + 128 + j] + bias_sum[128 + j];
        float gg = sm_g[m * 512 + 256 + j] + bias_sum[256 + j];
        float go = sm_g[m * 512 + 384 + j] + bias_sum[384 + j];
        float cn = sigmoidf_(gf) * sm_c[e] + sigmoidf_(gi) * tanhf(gg);
        sm_c[e] = cn;
        sm_h[e] = (_Float16)(sigmoidf_(go) * tanhf(cn));
      }
      __syncthreads();
    }
  }

  _Float16* sm_comb = sm_x;  // x is dead; combined is f16 (feeds WMMA only)
  float* sm_tag = sm_g;      // gates dead

  // combined = [h, emb[ident]] (16 x 256) @ W_comb^T + b_comb; wave w: tile w
  {
    v8f acc2[1] = {};
#pragma unroll
    for (int k0 = 0; k0 < 8; ++k0) {
      v16h a;
      if (k0 < 4) {
        a = frag_from_f16_lds(sm_h, 128, k0 * 32, lane);
      } else {
        int m = lane & 15;
        const float* erow = emb_table + (size_t)sm_ident[m] * 128 + (k0 - 4) * 32;
#pragma unroll
        for (int t = 0; t < 16; ++t) a[t] = (_Float16)erow[kmap(lane, t)];
      }
      acc2[0] = WMMA_F16(a, load_bfrag(Bcomb, wave * 8 + k0, lane), acc2[0]);
    }
    int col = lane & 15;
    int mb = (lane >> 4) << 3;
#pragma unroll
    for (int r = 0; r < 8; ++r)
      sm_comb[(mb + r) * 128 + wave * 16 + col] =
          (_Float16)(acc2[0][r] + b_comb[wave * 16 + col]);
  }
  __syncthreads();

  // tag = combined @ W_tag^T + b_tag; wave w: tile w
  {
    v8f acc3[1] = {};
#pragma unroll
    for (int k0 = 0; k0 < 4; ++k0) {
      v16h a = frag_from_f16_lds(sm_comb, 128, k0 * 32, lane);
      acc3[0] = WMMA_F16(a, load_bfrag(Btag, wave * 4 + k0, lane), acc3[0]);
    }
    int col = lane & 15;
    int mb = (lane >> 4) << 3;
#pragma unroll
    for (int r = 0; r < 8; ++r)
      sm_tag[(mb + r) * 128 + wave * 16 + col] =
          acc3[0][r] + b_tag[wave * 16 + col];
  }
  __syncthreads();

  // log_softmax over 128 tags: 16 lanes per node, shuffle-reduced.
  {
    const int m = tid >> 4;
    const int sub = tid & 15;
    float v[8];
    float mx = -3.0e38f;
#pragma unroll
    for (int k = 0; k < 8; ++k) {
      v[k] = sm_tag[m * 128 + sub + 16 * k];
      mx = fmaxf(mx, v[k]);
    }
    mx = fmaxf(mx, __shfl_xor(mx, 8, 16));
    mx = fmaxf(mx, __shfl_xor(mx, 4, 16));
    mx = fmaxf(mx, __shfl_xor(mx, 2, 16));
    mx = fmaxf(mx, __shfl_xor(mx, 1, 16));
    float ssum = 0.0f;
#pragma unroll
    for (int k = 0; k < 8; ++k) ssum += __expf(v[k] - mx);
    ssum += __shfl_xor(ssum, 8, 16);
    ssum += __shfl_xor(ssum, 4, 16);
    ssum += __shfl_xor(ssum, 2, 16);
    ssum += __shfl_xor(ssum, 1, 16);
    const float lz = __logf(ssum) + mx;
    const int node = base + m;
    if (node < N) {
#pragma unroll
      for (int k = 0; k < 8; ++k) {
        float r = v[k] - lz;
        int j = sub + 16 * k;
        scores_out[(size_t)node * 128 + j] = (_Float16)r;
        if (out_f32 != nullptr && node == 0) out_f32[j] = r;
      }
    }
  }
}

// ---------------------------------------------------------------------------
// Leaf gather: leaf scores depend only on ident -> broadcast 26-row table.
// ---------------------------------------------------------------------------
__global__ __launch_bounds__(256) void leaf_gather_kernel(
    _Float16* __restrict__ dst, const _Float16* __restrict__ table,
    const int* __restrict__ idents_leaf, int n_nodes) {
  const int4* tab4 = reinterpret_cast<const int4*>(table);
  int4* dst4 = reinterpret_cast<int4*>(dst);
  const size_t total = (size_t)n_nodes * 16;  // 16 int4 per node (256 B)
  for (size_t i = (size_t)blockIdx.x * blockDim.x + threadIdx.x; i < total;
       i += (size_t)gridDim.x * blockDim.x) {
    size_t node = i >> 4;
    int c = (int)(i & 15);
    dst4[node * 16 + c] = tab4[(size_t)idents_leaf[node] * 16 + c];
  }
}

// ---------------------------------------------------------------------------
// Host-side orchestration.
// ---------------------------------------------------------------------------
static constexpr int DEPTH_C = 17;

static constexpr size_t OFF_BIAS  = 0;                    // 512 f32
static constexpr size_t OFF_H1    = 2048;                 // 128 f32
static constexpr size_t OFF_C1    = 2560;                 // 128 f32
static constexpr size_t OFF_TABLE = 3072;                 // 26*128 f16
static constexpr size_t OFF_BIH   = 10240;                // 128 tiles * 1 KB
static constexpr size_t OFF_BHH   = OFF_BIH + 131072;
static constexpr size_t OFF_BCOMB = OFF_BHH + 131072;     // 64 tiles
static constexpr size_t OFF_BTAG  = OFF_BCOMB + 65536;    // 32 tiles
static constexpr size_t OFF_BUFA  = OFF_BTAG + 32768;     // 65536*128 f16
static constexpr size_t OFF_BUFB  = OFF_BUFA + 16777216;  // 32768*128 f16

extern "C" void kernel_launch(void* const* d_in, const int* in_sizes, int n_in,
                              void* d_out, int out_size, void* d_ws,
                              size_t ws_size, hipStream_t stream) {
  (void)in_sizes; (void)n_in; (void)out_size; (void)ws_size;

  const int*   idents    = (const int*)d_in[0];
  const float* emb_table = (const float*)d_in[1];
  const float* W_ih      = (const float*)d_in[2];
  const float* W_hh      = (const float*)d_in[3];
  const float* b_ih      = (const float*)d_in[4];
  const float* b_hh      = (const float*)d_in[5];
  const float* lstm_init = (const float*)d_in[6];
  const float* W_comb    = (const float*)d_in[7];
  const float* b_comb    = (const float*)d_in[8];
  const float* W_tag     = (const float*)d_in[9];
  const float* b_tag     = (const float*)d_in[10];
  float* out = (float*)d_out;

  char* ws = (char*)d_ws;
  float*    bias_sum = (float*)(ws + OFF_BIAS);
  float*    h1       = (float*)(ws + OFF_H1);
  float*    c1       = (float*)(ws + OFF_C1);
  _Float16* table    = (_Float16*)(ws + OFF_TABLE);
  _Float16* Bih      = (_Float16*)(ws + OFF_BIH);
  _Float16* Bhh      = (_Float16*)(ws + OFF_BHH);
  _Float16* Bcomb    = (_Float16*)(ws + OFF_BCOMB);
  _Float16* Btag     = (_Float16*)(ws + OFF_BTAG);
  _Float16* bufA     = (_Float16*)(ws + OFF_BUFA);
  _Float16* bufB     = (_Float16*)(ws + OFF_BUFB);

  prep_const_kernel<<<1, 512, 0, stream>>>(W_ih, b_ih, b_hh, lstm_init,
                                           bias_sum, h1, c1);
  prep_weights_kernel<<<352, 32, 0, stream>>>(W_ih, W_hh, W_comb, W_tag, Bih,
                                              Bhh, Bcomb, Btag);

  // Leaf table: 26 "nodes" with ident = iota, no children.
  level_kernel<<<2, 256, 0, stream>>>(
      nullptr, table, nullptr, 26, nullptr, Bih, Bhh, Bcomb, Btag, bias_sum,
      h1, c1, emb_table, b_comb, b_tag, /*has_children=*/0, /*use_iota=*/1);

  // Expand the table into the 65536-row leaf score buffer (level 16 -> bufA).
  {
    const int leafN = 1 << (DEPTH_C - 1);
    leaf_gather_kernel<<<1024, 256, 0, stream>>>(bufA, table,
                                                 idents + (leafN - 1), leafN);
  }

  // Internal levels 15 .. 0 (root). Even levels write bufA, odd write bufB.
  for (int d = DEPTH_C - 2; d >= 0; --d) {
    const int N = 1 << d;
    _Float16* dst = (d & 1) ? bufB : bufA;
    const _Float16* src = (d & 1) ? bufA : bufB;
    float* of = (d == 0) ? out : nullptr;
    const int blocks = (N + 15) >> 4;
    level_kernel<<<blocks, 256, 0, stream>>>(
        src, dst, of, N, idents + (N - 1), Bih, Bhh, Bcomb, Btag, bias_sum, h1,
        c1, emb_table, b_comb, b_tag, /*has_children=*/1, /*use_iota=*/0);
  }
}
